// CrossAttention_77386720739682
// MI455X (gfx1250) — matrix operand order repaired and checked by reading, not compiled, not run on hardware
//
#include <hip/hip_runtime.h>
#include <hip/hip_bf16.h>
#include <cstdint>

// ---------------------------------------------------------------------------
// CrossAttention for MI455X (gfx1250, wave32, WMMA bf16 16x16x32)
//   B=8, QL=1, KVL=4096, E=1024, H=16, D=64
// K-projection is folded into a tiny per-head vector u = Wk_head @ Qh, so the
// only big GEMMs are value@Wv and (V*aw)@Wo, fused into one LDS-tiled kernel
// whose value tile is staged with CDNA5 async global->LDS copies.
// ---------------------------------------------------------------------------

#define BB   8
#define KVL  4096
#define EE   1024
#define HH   16
#define DD   64

typedef __attribute__((ext_vector_type(16))) __bf16       v16bf;
typedef __attribute__((ext_vector_type(8)))  float        v8f;
typedef __attribute__((ext_vector_type(4)))  float        f32x4;
typedef __attribute__((ext_vector_type(4)))  unsigned int u32x4;
typedef __attribute__((ext_vector_type(2)))  unsigned int u32x2;

// exact parameter type of the async-LDS builtin (from hipcc diagnostic)
typedef int v4i_vs __attribute__((vector_size(4 * sizeof(int))));

#if __has_builtin(__builtin_amdgcn_global_load_async_to_lds_b128)
#define HAVE_ASYNC_LDS 1
#endif

static __device__ __forceinline__ void wait_async0() {
#if __has_builtin(__builtin_amdgcn_s_wait_asynccnt)
  __builtin_amdgcn_s_wait_asynccnt(0);
#else
  asm volatile("s_wait_asynccnt 0" ::: "memory");
#endif
}

// native f32 -> bf16 (RNE); lets clang use packed v_cvt on gfx1250
static __device__ __forceinline__ unsigned short f2bf(float f) {
  return __builtin_bit_cast(unsigned short, (__bf16)f);
}

union Frag16 {
  u32x4          q[2];
  unsigned short s[16];
  v16bf          v;
};

static __device__ __forceinline__ v8f wmma_bf16(const Frag16& a, const Frag16& b, v8f c) {
  return __builtin_amdgcn_wmma_f32_16x16x32_bf16(
      /*neg_a=*/false, a.v, /*neg_b=*/false, b.v,
      /*c_mod=*/(short)0, c, /*reuse_a=*/false, /*reuse_b=*/false);
}

// ---------------------------------------------------------------------------
// Kernel W: transpose + convert Wv, Wo (1024x1024 fp32) -> WvT, WoT (bf16)
// so GEMM B-fragments become two contiguous 16-byte loads per 16x32 tile.
// ---------------------------------------------------------------------------
__global__ __launch_bounds__(256) void ca_wconv(const float* __restrict__ Wv,
                                                const float* __restrict__ Wo,
                                                unsigned short* __restrict__ WvT,
                                                unsigned short* __restrict__ WoT) {
  const float*    src = blockIdx.y ? Wo : Wv;
  unsigned short* dst = blockIdx.y ? WoT : WvT;
  const int idx = (blockIdx.x * 256 + threadIdx.x) * 4;   // covers 1024*1024
  const int k = idx >> 10;
  const int n = idx & (EE - 1);
  const f32x4 v = *reinterpret_cast<const f32x4*>(src + idx);
  dst[(long)(n + 0) * EE + k] = f2bf(v.x);
  dst[(long)(n + 1) * EE + k] = f2bf(v.y);
  dst[(long)(n + 2) * EE + k] = f2bf(v.z);
  dst[(long)(n + 3) * EE + k] = f2bf(v.w);
}

// ---------------------------------------------------------------------------
// Kernel A: Q[b,e] = query[b,:] @ Wq[:,e] + bq[e]   (8 x 1024, tiny)
// ---------------------------------------------------------------------------
__global__ __launch_bounds__(256) void ca_qproj(const float* __restrict__ query,
                                                const float* __restrict__ Wq,
                                                const float* __restrict__ bq,
                                                float* __restrict__ Q) {
  const int b = blockIdx.x;
  const int e = threadIdx.x * 4;
  f32x4 acc = *reinterpret_cast<const f32x4*>(bq + e);
  for (int c = 0; c < EE; ++c) {
    const float q = query[(long)b * EE + c];
    const f32x4 w = *reinterpret_cast<const f32x4*>(Wq + (long)c * EE + e);
    acc.x += q * w.x; acc.y += q * w.y; acc.z += q * w.z; acc.w += q * w.w;
  }
  *reinterpret_cast<f32x4*>(Q + (long)b * EE + e) = acc;
}

// ---------------------------------------------------------------------------
// Kernel B: u[b,h,c] = sum_j Q[b, h*64+j] * Wk[c, h*64+j]   (bf16 output)
//           sconst[b,h] = sum_j bk[h*64+j] * Q[b, h*64+j]
// Folds the entire 69-GFLOP K projection into ~8M MACs.
// ---------------------------------------------------------------------------
__global__ __launch_bounds__(256) void ca_uproj(const float* __restrict__ Q,
                                                const float* __restrict__ Wk,
                                                const float* __restrict__ bk,
                                                unsigned short* __restrict__ u_bf,
                                                float* __restrict__ sconst) {
  __shared__ float qh[DD];
  __shared__ float red[DD];
  const int bh = blockIdx.x;           // b*16 + h
  const int b = bh >> 4, h = bh & 15;
  const int t = threadIdx.x;
  if (t < DD) {
    const float q = Q[(long)b * EE + h * DD + t];
    qh[t]  = q;
    red[t] = q * bk[h * DD + t];
  }
  __syncthreads();
  for (int off = 32; off >= 1; off >>= 1) {
    if (t < off) red[t] += red[t + off];
    __syncthreads();
  }
  if (t == 0) sconst[bh] = red[0];

  const int c0 = t * 4;
  #pragma unroll
  for (int r = 0; r < 4; ++r) {
    const float* wp = Wk + (long)(c0 + r) * EE + h * DD;
    float a = 0.0f;
    for (int j = 0; j < DD; ++j) a += qh[j] * wp[j];
    u_bf[(long)bh * EE + c0 + r] = f2bf(a);
  }
}

// ---------------------------------------------------------------------------
// Kernel C: scores[b,h,k] = (key[b,k,:] . u[b,h,:] + sconst[b,h]) / 8, masked.
// WMMA: M=16 key rows, N=16 heads, K=1024 (32 bf16 k-steps). One wave / tile.
// ---------------------------------------------------------------------------
__global__ __launch_bounds__(256) void ca_scores(const float* __restrict__ key,
                                                 const unsigned short* __restrict__ u_bf,
                                                 const float* __restrict__ sconst,
                                                 const int* __restrict__ mask,
                                                 float* __restrict__ scores) {
  const int t  = threadIdx.x;
  const int w  = t >> 5;
  const int l  = t & 31;
  const int hH = l >> 4;
  const int ln = l & 15;
  const int b  = blockIdx.x >> 5;                 // 32 blocks per batch
  const int r0 = (blockIdx.x & 31) * 128 + w * 16;

  v8f acc;
  #pragma unroll
  for (int i = 0; i < 8; ++i) acc[i] = 0.0f;

  for (int ks = 0; ks < 32; ++ks) {
    // A fragment: key rows (fp32 -> bf16)
    Frag16 a;
    const float* kp = key + ((long)(b * KVL + r0 + ln)) * EE + ks * 32 + hH * 8;
    const f32x4 f0 = *reinterpret_cast<const f32x4*>(kp);
    const f32x4 f1 = *reinterpret_cast<const f32x4*>(kp + 4);
    const f32x4 f2 = *reinterpret_cast<const f32x4*>(kp + 16);
    const f32x4 f3 = *reinterpret_cast<const f32x4*>(kp + 20);
    a.s[0]=f2bf(f0.x); a.s[1]=f2bf(f0.y); a.s[2]=f2bf(f0.z); a.s[3]=f2bf(f0.w);
    a.s[4]=f2bf(f1.x); a.s[5]=f2bf(f1.y); a.s[6]=f2bf(f1.z); a.s[7]=f2bf(f1.w);
    a.s[8]=f2bf(f2.x); a.s[9]=f2bf(f2.y); a.s[10]=f2bf(f2.z); a.s[11]=f2bf(f2.w);
    a.s[12]=f2bf(f3.x); a.s[13]=f2bf(f3.y); a.s[14]=f2bf(f3.z); a.s[15]=f2bf(f3.w);
    // B fragment: u (head = N lane)
    Frag16 bf;
    const unsigned short* up = u_bf + ((long)(b * HH + ln)) * EE + ks * 32 + hH * 8;
    bf.q[0] = *reinterpret_cast<const u32x4*>(up);
    bf.q[1] = *reinterpret_cast<const u32x4*>(up + 16);
    acc = wmma_bf16(a, bf, acc);
  }

  const int h  = ln;
  const float sc = sconst[b * HH + h];
  #pragma unroll
  for (int i = 0; i < 8; ++i) {
    const int k = r0 + hH * 8 + i;
    float val = (acc[i] + sc) * 0.125f;          // 1/sqrt(64)
    if (mask[(long)b * KVL + k] == 0) val = -__builtin_inff();
    scores[((long)(b * HH + h) << 12) + k] = val;
  }
}

// ---------------------------------------------------------------------------
// Kernel D: softmax over KVL per (b,h); writes aw (ws) and the broadcast
// aw_exp output (B,H,KVL,64) — 134 MB of coalesced float4 stores.
// ---------------------------------------------------------------------------
__global__ __launch_bounds__(256) void ca_softmax(const float* __restrict__ scores,
                                                  float* __restrict__ aw,
                                                  float* __restrict__ aw_exp_out) {
  __shared__ float red[256];
  const int bh = blockIdx.x;
  const int t  = threadIdx.x;
  const float* s = scores + ((long)bh << 12);

  float v[16];
  float m = -__builtin_inff();
  #pragma unroll
  for (int i = 0; i < 16; ++i) { v[i] = s[t + i * 256]; m = fmaxf(m, v[i]); }
  red[t] = m; __syncthreads();
  for (int off = 128; off >= 1; off >>= 1) {
    if (t < off) red[t] = fmaxf(red[t], red[t + off]);
    __syncthreads();
  }
  m = red[0]; __syncthreads();

  float sum = 0.0f;
  #pragma unroll
  for (int i = 0; i < 16; ++i) { v[i] = __expf(v[i] - m); sum += v[i]; }
  red[t] = sum; __syncthreads();
  for (int off = 128; off >= 1; off >>= 1) {
    if (t < off) red[t] += red[t + off];
    __syncthreads();
  }
  const float inv = 1.0f / red[0];

  #pragma unroll
  for (int i = 0; i < 16; ++i) {
    const int k = t + i * 256;
    const float a = v[i] * inv;
    aw[((long)bh << 12) + k] = a;
    f32x4 a4; a4.x = a; a4.y = a; a4.z = a; a4.w = a;
    f32x4* dst = reinterpret_cast<f32x4*>(aw_exp_out + (((long)bh << 12) + k) * DD);
    #pragma unroll
    for (int j = 0; j < 16; ++j) dst[j] = a4;
  }
}

// ---------------------------------------------------------------------------
// Kernel E (the big one): per 32-row tile (flattened b*KVL+k):
//   stage  : value tile (32x1024 fp32, 128 KB) -> LDS via ASYNC global->LDS
//   phase 1: V = value_tile @ Wv            (bf16 WMMA, acc f32)
//   scale  : V = (V + bv) * aw[b,h(col),k]  -> bf16 tile aliasing LDS front half
//   phase 2: out = V_scaled @ Wo + bo       (bf16 WMMA, acc f32)
// 8 waves/block; wave owns a 128-col slab (8 N-tiles) x 2 M-tiles.
// 137 GFLOP across 1024 blocks; bf16 weights (2 MB each) stream from L2.
// ---------------------------------------------------------------------------
__global__ __launch_bounds__(256) void ca_fused_vo(const float* __restrict__ value,
                                                   const float* __restrict__ bv,
                                                   const float* __restrict__ bo,
                                                   const unsigned short* __restrict__ WvT,
                                                   const unsigned short* __restrict__ WoT,
                                                   const float* __restrict__ aw,
                                                   float* __restrict__ out) {
  __shared__ float fval[32 * EE];                  // 128 KB fp32 value tile
  unsigned short* tile = reinterpret_cast<unsigned short*>(fval); // 64 KB alias
  const int  t     = threadIdx.x;
  const int  w     = t >> 5;
  const int  l     = t & 31;
  const int  hH    = l >> 4;
  const int  ln    = l & 15;
  const long grow0 = (long)blockIdx.x * 32;        // flattened row base
  const int  b     = (int)(grow0 >> 12);
  const int  k0    = (int)(grow0 & (KVL - 1));

  // ---- stage 32x1024 fp32 value tile into LDS ----
#ifdef HAVE_ASYNC_LDS
  #pragma unroll
  for (int i = 0; i < 32; ++i) {
    const int e = (i * 256 + t) * 4;               // float index, 16B chunks
    const float* gp = value + grow0 * EE + e;
    __builtin_amdgcn_global_load_async_to_lds_b128(
        (v4i_vs*)gp,
        (__attribute__((address_space(3))) v4i_vs*)&fval[e], 0, 0);
  }
  wait_async0();
#else
  #pragma unroll
  for (int i = 0; i < 32; ++i) {
    const int e = (i * 256 + t) * 4;
    *reinterpret_cast<f32x4*>(&fval[e]) =
        *reinterpret_cast<const f32x4*>(value + grow0 * EE + e);
  }
#endif
  __syncthreads();

  v8f acc[2][8];
  #pragma unroll
  for (int m = 0; m < 2; ++m)
    #pragma unroll
    for (int n = 0; n < 8; ++n)
      #pragma unroll
      for (int i = 0; i < 8; ++i) acc[m][n][i] = 0.0f;

  // -------- phase 1: value_tile @ Wv (A gathered fp32 from LDS, cvt bf16) ---
  for (int ks = 0; ks < 32; ++ks) {
    Frag16 afr[2];
    #pragma unroll
    for (int m = 0; m < 2; ++m) {
      const float* ap = &fval[(m * 16 + ln) * EE + ks * 32 + hH * 8];
      const f32x4 g0 = *reinterpret_cast<const f32x4*>(ap);        // K h*8+0..3
      const f32x4 g1 = *reinterpret_cast<const f32x4*>(ap + 4);    // K h*8+4..7
      const f32x4 g2 = *reinterpret_cast<const f32x4*>(ap + 16);   // K h*8+16..19
      const f32x4 g3 = *reinterpret_cast<const f32x4*>(ap + 20);   // K h*8+20..23
      afr[m].s[0]=f2bf(g0.x);  afr[m].s[1]=f2bf(g0.y);  afr[m].s[2]=f2bf(g0.z);  afr[m].s[3]=f2bf(g0.w);
      afr[m].s[4]=f2bf(g1.x);  afr[m].s[5]=f2bf(g1.y);  afr[m].s[6]=f2bf(g1.z);  afr[m].s[7]=f2bf(g1.w);
      afr[m].s[8]=f2bf(g2.x);  afr[m].s[9]=f2bf(g2.y);  afr[m].s[10]=f2bf(g2.z); afr[m].s[11]=f2bf(g2.w);
      afr[m].s[12]=f2bf(g3.x); afr[m].s[13]=f2bf(g3.y); afr[m].s[14]=f2bf(g3.z); afr[m].s[15]=f2bf(g3.w);
    }
    #pragma unroll
    for (int n8 = 0; n8 < 8; ++n8) {
      const int ncol = w * 128 + n8 * 16 + ln;
      const unsigned short* bp = WvT + (long)ncol * EE + ks * 32 + hH * 8;
      Frag16 bfr;
      bfr.q[0] = *reinterpret_cast<const u32x4*>(bp);
      bfr.q[1] = *reinterpret_cast<const u32x4*>(bp + 16);
      acc[0][n8] = wmma_bf16(afr[0], bfr, acc[0][n8]);
      acc[1][n8] = wmma_bf16(afr[1], bfr, acc[1][n8]);
    }
  }
  __syncthreads();            // all waves finished reading the fp32 tile

  // -------- scale by attention weight, + bv, to bf16 tile (LDS alias) ------
  #pragma unroll
  for (int m = 0; m < 2; ++m) {
    #pragma unroll
    for (int n8 = 0; n8 < 8; ++n8) {
      const int ncol = w * 128 + n8 * 16 + ln;
      const int h    = ncol >> 6;
      const float bvv = bv[ncol];
      const float* awp = aw + ((long)(b * HH + h) << 12) + k0;
      #pragma unroll
      for (int i = 0; i < 8; ++i) {
        const int row = m * 16 + hH * 8 + i;
        const float val = (acc[m][n8][i] + bvv) * awp[row];
        tile[row * EE + ncol] = f2bf(val);
      }
    }
  }
  __syncthreads();

  #pragma unroll
  for (int m = 0; m < 2; ++m)
    #pragma unroll
    for (int n = 0; n < 8; ++n)
      #pragma unroll
      for (int i = 0; i < 8; ++i) acc[m][n][i] = 0.0f;

  // -------- phase 2: scaled bf16 tile @ Wo --------
  for (int ks = 0; ks < 32; ++ks) {
    Frag16 afr[2];
    #pragma unroll
    for (int m = 0; m < 2; ++m) {
      const unsigned short* ap = &tile[(m * 16 + ln) * EE + ks * 32 + hH * 8];
      afr[m].q[0] = *reinterpret_cast<const u32x4*>(ap);        // K h*8 .. +7
      afr[m].q[1] = *reinterpret_cast<const u32x4*>(ap + 16);   // K h*8+16 .. +23
    }
    #pragma unroll
    for (int n8 = 0; n8 < 8; ++n8) {
      const int ncol = w * 128 + n8 * 16 + ln;
      const unsigned short* bp = WoT + (long)ncol * EE + ks * 32 + hH * 8;
      Frag16 bfr;
      bfr.q[0] = *reinterpret_cast<const u32x4*>(bp);
      bfr.q[1] = *reinterpret_cast<const u32x4*>(bp + 16);
      acc[0][n8] = wmma_bf16(afr[0], bfr, acc[0][n8]);
      acc[1][n8] = wmma_bf16(afr[1], bfr, acc[1][n8]);
    }
  }

  // -------- epilogue: + bo, fp32 store (coalesced 64B per half-wave row) ----
  #pragma unroll
  for (int m = 0; m < 2; ++m) {
    #pragma unroll
    for (int n8 = 0; n8 < 8; ++n8) {
      const int ncol = w * 128 + n8 * 16 + ln;
      const float bov = bo[ncol];
      #pragma unroll
      for (int i = 0; i < 8; ++i) {
        const int row = m * 16 + hH * 8 + i;
        out[(grow0 + row) * EE + ncol] = acc[m][n8][i] + bov;
      }
    }
  }
}

// ---------------------------------------------------------------------------
extern "C" void kernel_launch(void* const* d_in, const int* in_sizes, int n_in,
                              void* d_out, int out_size, void* d_ws, size_t ws_size,
                              hipStream_t stream) {
  (void)in_sizes; (void)n_in; (void)out_size; (void)ws_size;
  const float* query = (const float*)d_in[0];
  const float* key   = (const float*)d_in[1];
  const float* value = (const float*)d_in[2];
  const int*   mask  = (const int*)d_in[3];
  const float* Wq    = (const float*)d_in[4];
  const float* bq    = (const float*)d_in[5];
  const float* Wk    = (const float*)d_in[6];
  const float* bk    = (const float*)d_in[7];
  const float* Wv    = (const float*)d_in[8];
  const float* bv    = (const float*)d_in[9];
  const float* Wo    = (const float*)d_in[10];
  const float* bo    = (const float*)d_in[11];

  float* out        = (float*)d_out;
  float* aw_exp_out = out + (long)BB * KVL * EE;   // second tuple element

  // workspace layout (~8.5 MB total)
  char* ws = (char*)d_ws;
  float*          Qp     = (float*)(ws);                                   // 32 KB
  float*          sconst = (float*)(ws + (64l  << 10));                    // 512 B
  unsigned short* u_bf   = (unsigned short*)(ws + (128l << 10));           // 256 KB
  float*          scores = (float*)(ws + (512l << 10));                    // 2 MB
  float*          aw     = (float*)(ws + (512l << 10) + (2l << 20));       // 2 MB
  unsigned short* WvT    = (unsigned short*)(ws + (512l << 10) + (4l << 20)); // 2 MB
  unsigned short* WoT    = WvT + 1024 * 1024;                              // 2 MB

  ca_wconv  <<<dim3(1024, 2),        256, 0, stream>>>(Wv, Wo, WvT, WoT);
  ca_qproj  <<<BB,                   256, 0, stream>>>(query, Wq, bq, Qp);
  ca_uproj  <<<BB * HH,              256, 0, stream>>>(Qp, Wk, bk, u_bf, sconst);
  ca_scores <<<BB * (KVL / 128),     256, 0, stream>>>(key, u_bf, sconst, mask, scores);
  ca_softmax<<<BB * HH,              256, 0, stream>>>(scores, aw, aw_exp_out);
  ca_fused_vo<<<(BB * KVL) / 32,     256, 0, stream>>>(value, bv, bo, WvT, WoT, aw, out);
}
